// GraphModel_51874615001237
// MI455X (gfx1250) — matrix operand order repaired and checked by reading, not compiled, not run on hardware
//
#include <hip/hip_runtime.h>
#include <hip/hip_bf16.h>
#include <math.h>

// ---------------- problem constants (match reference) ----------------
#define BB 32
#define TT 30
#define VV 20000
#define HH 512
#define FPP 512
#define NFF 26
#define NOO 20
#define START_ID 1

typedef __bf16 bf16;
typedef __attribute__((ext_vector_type(16))) __bf16 v16bf;
typedef __attribute__((ext_vector_type(8)))  __bf16 v8bf;
typedef __attribute__((ext_vector_type(8)))  float  v8f;

__device__ __forceinline__ float sigf(float x) { return 1.0f / (1.0f + __expf(-x)); }

// ---------------------------------------------------------------------
// bf16 WMMA GEMM:  C[M,N] = A[M,K] @ W[N,K]^T + bias, optional tanh,
// optional bf16 shadow copy of C.
// One wave computes a 16 x (16*NT) strip: the A tile is loaded ONCE per
// K-step and reused across NT accumulators (cuts A traffic NT-fold; the
// vocab GEMM is L2-bandwidth bound on A with NT=1).
// Operand layouts used here (contiguous k order, not the ISA table order; valid as A and B match):
//   A 16x32 bf16: lanes 0-15 = M rows, K(0..15) contiguous;
//                 lanes 16-31 same rows, K(16..31).
//   B 32x16 bf16: lane n (0-15) holds column n, K(0..15) contiguous;
//                 lanes 16-31 hold K(16..31).
// Both map to contiguous b128 loads from row-major [rows,K] arrays.
// ---------------------------------------------------------------------
template <int NT>
__global__ void gemm_bf16(const bf16* __restrict__ A, const bf16* __restrict__ W,
                          const float* __restrict__ bias,
                          float* __restrict__ C, bf16* __restrict__ Cbf,
                          int M, int N, int K, int ldc, int act)
{
    const int lane   = threadIdx.x & 31;
    const int wave   = threadIdx.x >> 5;
    const int tilesN = N >> 4;
    const int gN     = tilesN / NT;                 // N-groups
    const int groups = (M >> 4) * gN;
    const int grp    = blockIdx.x * (blockDim.x >> 5) + wave;
    if (grp >= groups) return;                      // wave-uniform exit (EXEC all-ones below)

    const int m0   = (grp / gN) << 4;
    const int n0g  = (grp % gN) * (16 * NT);
    const int half = lane >> 4;                     // 0 | 1
    const int l16  = lane & 15;

    const bf16* arow  = A + (size_t)(m0 + l16) * K + (half ? 16 : 0);
    const bf16* bbase = W + (size_t)(n0g + l16) * K + (half ? 16 : 0);

    v8f acc[NT];
#pragma unroll
    for (int nt = 0; nt < NT; ++nt) acc[nt] = (v8f){};

    for (int k0 = 0; k0 < K; k0 += 32) {
        v16bf a = *(const v16bf*)(arow + k0);
#pragma unroll
        for (int nt = 0; nt < NT; ++nt) {
            v16bf b = *(const v16bf*)(bbase + (size_t)nt * 16 * K + k0);
            acc[nt] = __builtin_amdgcn_wmma_f32_16x16x32_bf16(false, a, false, b,
                                                              (short)0, acc[nt], false, false);
        }
    }

#pragma unroll
    for (int nt = 0; nt < NT; ++nt) {
        const int   n  = n0g + nt * 16 + l16;
        const float bn = bias ? bias[n] : 0.0f;
#pragma unroll
        for (int r = 0; r < 8; ++r) {
            const int m = m0 + r + (half ? 8 : 0);
            float v = acc[nt][r] + bn;
            if (act == 1) v = tanhf(v);
            C[(size_t)m * ldc + n] = v;
            if (Cbf) Cbf[(size_t)m * ldc + n] = (bf16)v;
        }
    }
}

// ---------------- small helper kernels ----------------
__global__ void k_f32_to_bf16(const float* __restrict__ s, bf16* __restrict__ d, long n) {
    long i = (long)blockIdx.x * blockDim.x + threadIdx.x;
    if (i < n) d[i] = (bf16)s[i];
}

__global__ void k_fill0(float* p, int n) {
    int i = blockIdx.x * blockDim.x + threadIdx.x;
    if (i < n) p[i] = 0.0f;
}

// Wcat[row, 0:K1] = w1[row,:], Wcat[row, K1:K1+K2] = w2[row,:]  (f32 -> bf16)
__global__ void k_cat_weights(const float* __restrict__ w1, int K1,
                              const float* __restrict__ w2, int K2,
                              bf16* __restrict__ dst) {
    const int row = blockIdx.x;
    const int K = K1 + K2;
    for (int k = threadIdx.x; k < K; k += blockDim.x) {
        float v = (k < K1) ? w1[(size_t)row * K1 + k] : w2[(size_t)row * K2 + (k - K1)];
        dst[(size_t)row * K + k] = (bf16)v;
    }
}

// g_feat[b] = [mean_f frame(1024) , sum_o obj / sum(mask) (512)]
__global__ void k_gfeat(const float* __restrict__ frame, const float* __restrict__ obj,
                        const float* __restrict__ mask, float* __restrict__ g,
                        bf16* __restrict__ gbf) {
    const int b = blockIdx.x;
    float msum = 0.0f;
    for (int o = 0; o < NOO; ++o) msum += mask[b * NOO + o];
    for (int d = threadIdx.x; d < 3 * FPP; d += blockDim.x) {
        float acc = 0.0f;
        if (d < 2 * FPP) {
            for (int f = 0; f < NFF; ++f) acc += frame[((size_t)b * NFF + f) * (2 * FPP) + d];
            acc *= (1.0f / NFF);
        } else {
            const int dd = d - 2 * FPP;
            for (int o = 0; o < NOO; ++o) acc += obj[((size_t)b * NOO + o) * FPP + dd];
            acc /= msum;
        }
        g[(size_t)b * (3 * FPP) + d] = acc;
        gbf[(size_t)b * (3 * FPP) + d] = (bf16)acc;
    }
}

// xh_att[b] = [lang_h(512), word(512), g_feat(1536), att_h(512)]  (bf16)
// t < 0: init mode (lang_h = att_h = 0, word = init_word); t == 0: start word
__global__ void k_pack_att(const float* __restrict__ lang_h, const float* __restrict__ att_h,
                           const float* __restrict__ g_feat, const float* __restrict__ init_word,
                           const float* __restrict__ word_embed, const int* __restrict__ captions,
                           int t, bf16* __restrict__ xh) {
    const int b = blockIdx.x;
    const float* wsrc;
    if (t < 0) wsrc = init_word + (size_t)b * HH;
    else {
        const int idx = (t == 0) ? START_ID : captions[b * TT + (t - 1)];
        wsrc = word_embed + (size_t)idx * HH;
    }
    for (int k = threadIdx.x; k < 3072; k += blockDim.x) {
        float v;
        if (k < 512)        v = (t < 0) ? 0.0f : lang_h[(size_t)b * HH + k];
        else if (k < 1024)  v = wsrc[k - 512];
        else if (k < 2560)  v = g_feat[(size_t)b * 1536 + (k - 1024)];
        else                v = (t < 0) ? 0.0f : att_h[(size_t)b * HH + (k - 2560)];
        xh[(size_t)b * 3072 + k] = (bf16)v;
    }
}

// gates [B,2048] (i,f,g,o), in-place c/h update, optional bf16 copy of h
__global__ void k_lstm_pw(const float* __restrict__ gates, float* __restrict__ h,
                          float* __restrict__ c, bf16* __restrict__ hbf) {
    const int i = blockIdx.x * blockDim.x + threadIdx.x;
    if (i >= BB * HH) return;
    const int b = i >> 9, d = i & 511;
    const float* g = gates + (size_t)b * 4 * HH;
    const float ig = sigf(g[d]);
    const float fg = sigf(g[HH + d]);
    const float gg = tanhf(g[2 * HH + d]);
    const float og = sigf(g[3 * HH + d]);
    const float cn = fg * c[i] + ig * gg;
    const float hn = og * tanhf(cn);
    c[i] = cn; h[i] = hn;
    if (hbf) hbf[i] = (bf16)hn;
}

// soft attention: alpha_n = softmax(tanh(q + v_pre[n]) . wa + wa_b), ctx = sum alpha_n feats[n]
__global__ void k_attn(const float* __restrict__ feats, const float* __restrict__ vpre,
                       const float* __restrict__ q, const float* __restrict__ wa,
                       const float* __restrict__ wa_b, const float* __restrict__ mask,
                       int Nn, int D, float* __restrict__ ctx) {
    __shared__ float salpha[32];
    __shared__ float red[256];
    const int b = blockIdx.x, tid = threadIdx.x;
    const float* qp = q + (size_t)b * 512;
    for (int n = 0; n < Nn; ++n) {
        const float* vp = vpre + ((size_t)b * Nn + n) * 512;
        float p = 0.0f;
        for (int d = tid; d < 512; d += 256) p += tanhf(qp[d] + vp[d]) * wa[d];
        red[tid] = p; __syncthreads();
        for (int s = 128; s > 0; s >>= 1) { if (tid < s) red[tid] += red[tid + s]; __syncthreads(); }
        if (tid == 0) {
            float a = red[0] + wa_b[0];
            if (mask && mask[b * Nn + n] == 0.0f) a = -1e9f;
            salpha[n] = a;
        }
        __syncthreads();
    }
    if (tid == 0) {
        float mx = salpha[0];
        for (int n = 1; n < Nn; ++n) mx = fmaxf(mx, salpha[n]);
        float s = 0.0f;
        for (int n = 0; n < Nn; ++n) { salpha[n] = __expf(salpha[n] - mx); s += salpha[n]; }
        const float inv = 1.0f / s;
        for (int n = 0; n < Nn; ++n) salpha[n] *= inv;
    }
    __syncthreads();
    for (int d = tid; d < D; d += 256) {
        float acc = 0.0f;
        for (int n = 0; n < Nn; ++n) acc += salpha[n] * feats[((size_t)b * Nn + n) * D + d];
        ctx[(size_t)b * D + d] = acc;
    }
}

// loc[b] = [obj_ctx(512), frm_ctx(1024)]  (bf16)
__global__ void k_pack_loc(const float* __restrict__ oc, const float* __restrict__ fc,
                           bf16* __restrict__ loc) {
    const int b = blockIdx.x;
    for (int k = threadIdx.x; k < 1536; k += blockDim.x) {
        float v = (k < 512) ? oc[(size_t)b * 512 + k] : fc[(size_t)b * 1024 + (k - 512)];
        loc[(size_t)b * 1536 + k] = (bf16)v;
    }
}

// dst[b] = [a(512), b2(512), c2(512)]  (bf16)
__global__ void k_pack3(const float* __restrict__ a, const float* __restrict__ b2,
                        const float* __restrict__ c2, bf16* __restrict__ dst) {
    const int b = blockIdx.x;
    for (int k = threadIdx.x; k < 1536; k += blockDim.x) {
        float v = (k < 512) ? a[(size_t)b * 512 + k]
                : (k < 1024) ? b2[(size_t)b * 512 + (k - 512)]
                             : c2[(size_t)b * 512 + (k - 1024)];
        dst[(size_t)b * 1536 + k] = (bf16)v;
    }
}

// ---------------------------------------------------------------------
extern "C" void kernel_launch(void* const* d_in, const int* in_sizes, int n_in,
                              void* d_out, int out_size, void* d_ws, size_t ws_size,
                              hipStream_t stream) {
    (void)in_sizes; (void)n_in; (void)out_size; (void)ws_size;

    // inputs (setup_inputs order)
    const float* frame_feats = (const float*)d_in[0];   // [B,NF,1024]
    const float* object_feats= (const float*)d_in[1];   // [B,NO,512]
    const float* object_masks= (const float*)d_in[2];   // [B,NO]
    const int*   captions    = (const int*)  d_in[3];   // [B,T]
    const float* word_embed  = (const float*)d_in[4];   // [V,512]
    const float* lstm_init_w = (const float*)d_in[5];   // [512,1536]
    const float* lstm_init_b = (const float*)d_in[6];   // [512]
    const float* att_ih      = (const float*)d_in[7];   // [2048,2560]
    const float* att_hh      = (const float*)d_in[8];   // [2048,512]
    const float* att_b       = (const float*)d_in[9];   // [2048]
    const float* lang_ih     = (const float*)d_in[10];  // [2048,1024]
    const float* lang_hh     = (const float*)d_in[11];  // [2048,512]
    const float* lang_b      = (const float*)d_in[12];  // [2048]
    const float* reg_wv      = (const float*)d_in[13];  // [512,512]
    const float* reg_wv_b    = (const float*)d_in[14];
    const float* reg_wh      = (const float*)d_in[15];  // [512,512]
    const float* reg_wh_b    = (const float*)d_in[16];
    const float* reg_wa      = (const float*)d_in[17];  // [512]
    const float* reg_wa_b    = (const float*)d_in[18];  // scalar
    const float* frm_wv      = (const float*)d_in[19];  // [512,1024]
    const float* frm_wv_b    = (const float*)d_in[20];
    const float* frm_wh      = (const float*)d_in[21];  // [512,512]
    const float* frm_wh_b    = (const float*)d_in[22];
    const float* frm_wa      = (const float*)d_in[23];  // [512]
    const float* frm_wa_b    = (const float*)d_in[24];  // scalar
    const float* fuse_w      = (const float*)d_in[25];  // [512,1536]
    const float* fuse_b      = (const float*)d_in[26];
    const float* out_w       = (const float*)d_in[27];  // [512,1536]
    const float* out_b       = (const float*)d_in[28];
    const float* restore_w   = (const float*)d_in[29];  // [V,512]
    const float* restore_b   = (const float*)d_in[30];

    float* out = (float*)d_out;                         // [B,T,V] f32

    // ---------------- workspace carve-out ----------------
    char* ws = (char*)d_ws;
    size_t off = 0;
    auto alloc = [&](size_t bytes) -> char* {
        off = (off + 255) & ~(size_t)255;
        char* p = ws + off; off += bytes; return p;
    };
    // bf16 weights
    bf16* watt   = (bf16*)alloc((size_t)2048 * 3072 * 2);   // [att_ih|att_hh]
    bf16* wlang  = (bf16*)alloc((size_t)2048 * 1536 * 2);   // [lang_ih|lang_hh]
    bf16* wres   = (bf16*)alloc((size_t)VV * 512 * 2);
    bf16* wout   = (bf16*)alloc((size_t)512 * 1536 * 2);
    bf16* wfuse  = (bf16*)alloc((size_t)512 * 1536 * 2);
    bf16* wregv  = (bf16*)alloc((size_t)512 * 512 * 2);
    bf16* wfrmv  = (bf16*)alloc((size_t)512 * 1024 * 2);
    bf16* wregh  = (bf16*)alloc((size_t)512 * 512 * 2);
    bf16* wfrmh  = (bf16*)alloc((size_t)512 * 512 * 2);
    bf16* winit  = (bf16*)alloc((size_t)512 * 1536 * 2);
    bf16* objbf  = (bf16*)alloc((size_t)BB * NOO * 512 * 2);
    bf16* frmbf  = (bf16*)alloc((size_t)BB * NFF * 1024 * 2);
    // bf16 activations
    bf16* gbf    = (bf16*)alloc((size_t)BB * 1536 * 2);
    bf16* xh_att = (bf16*)alloc((size_t)BB * 3072 * 2);
    bf16* xh_lng = (bf16*)alloc((size_t)BB * 1536 * 2);
    bf16* locbf  = (bf16*)alloc((size_t)BB * 1536 * 2);
    bf16* deccat = (bf16*)alloc((size_t)BB * 1536 * 2);
    bf16* atthbf = (bf16*)alloc((size_t)BB * 512 * 2);
    bf16* dobf   = (bf16*)alloc((size_t)BB * 512 * 2);
    // f32 state / scratch
    float* gfeat = (float*)alloc((size_t)BB * 1536 * 4);
    float* iword = (float*)alloc((size_t)BB * 512 * 4);
    float* vobj  = (float*)alloc((size_t)BB * NOO * 512 * 4);
    float* vfrm  = (float*)alloc((size_t)BB * NFF * 512 * 4);
    float* gates = (float*)alloc((size_t)BB * 2048 * 4);
    float* atth  = (float*)alloc((size_t)BB * 512 * 4);
    float* attc  = (float*)alloc((size_t)BB * 512 * 4);
    float* lngh  = (float*)alloc((size_t)BB * 512 * 4);
    float* lngc  = (float*)alloc((size_t)BB * 512 * 4);
    float* qobj  = (float*)alloc((size_t)BB * 512 * 4);
    float* qfrm  = (float*)alloc((size_t)BB * 512 * 4);
    float* cobj  = (float*)alloc((size_t)BB * 512 * 4);
    float* cfrm  = (float*)alloc((size_t)BB * 1024 * 4);
    float* featv = (float*)alloc((size_t)BB * 512 * 4);
    float* decot = (float*)alloc((size_t)BB * 512 * 4);

    auto cvt = [&](const float* s, bf16* d, long n) {
        k_f32_to_bf16<<<(unsigned)((n + 255) / 256), 256, 0, stream>>>(s, d, n);
    };
    // small GEMMs: 1 tile / wave (max parallelism, latency-bound)
    auto gemm = [&](const bf16* A, const bf16* W, const float* bias, float* C, bf16* Cbf,
                    int M, int N, int K, int ldc, int act) {
        int tiles = (M / 16) * (N / 16);
        gemm_bf16<1><<<(tiles + 3) / 4, 128, 0, stream>>>(A, W, bias, C, Cbf, M, N, K, ldc, act);
    };
    // vocab GEMM: 5 N-tiles / wave, A loaded once per K-step (L2-traffic bound)
    auto gemm5 = [&](const bf16* A, const bf16* W, const float* bias, float* C,
                     int M, int N, int K, int ldc) {
        int groups = (M / 16) * (N / 16 / 5);
        gemm_bf16<5><<<(groups + 3) / 4, 128, 0, stream>>>(A, W, bias, C, nullptr, M, N, K, ldc, 0);
    };

    // ---- once per call: weight conversion / packing ----
    cvt(restore_w, wres, (long)VV * 512);
    cvt(out_w,  wout,  512L * 1536);
    cvt(fuse_w, wfuse, 512L * 1536);
    cvt(reg_wv, wregv, 512L * 512);
    cvt(frm_wv, wfrmv, 512L * 1024);
    cvt(reg_wh, wregh, 512L * 512);
    cvt(frm_wh, wfrmh, 512L * 512);
    cvt(lstm_init_w, winit, 512L * 1536);
    cvt(object_feats, objbf, (long)BB * NOO * 512);
    cvt(frame_feats,  frmbf, (long)BB * NFF * 1024);
    k_cat_weights<<<2048, 256, 0, stream>>>(att_ih, 2560, att_hh, 512, watt);
    k_cat_weights<<<2048, 256, 0, stream>>>(lang_ih, 1024, lang_hh, 512, wlang);
    k_fill0<<<(BB * 512 + 255) / 256, 256, 0, stream>>>(attc, BB * 512);
    k_fill0<<<(BB * 512 + 255) / 256, 256, 0, stream>>>(lngc, BB * 512);
    k_fill0<<<(BB * 512 + 255) / 256, 256, 0, stream>>>(lngh, BB * 512);

    // ---- global features, init word, hoisted attention value projections ----
    k_gfeat<<<BB, 256, 0, stream>>>(frame_feats, object_feats, object_masks, gfeat, gbf);
    gemm(gbf,   winit, lstm_init_b, iword, nullptr, BB, 512, 1536, 512, 1);   // tanh
    gemm(objbf, wregv, reg_wv_b,    vobj,  nullptr, BB * NOO, 512, 512, 512, 0);
    gemm(frmbf, wfrmv, frm_wv_b,    vfrm,  nullptr, BB * NFF, 512, 1024, 512, 0);

    // ---- initial recurrence (builds the scan carry) ----
    k_pack_att<<<BB, 256, 0, stream>>>(lngh, atth, gfeat, iword, word_embed, captions, -1, xh_att);
    gemm(xh_att, watt, att_b, gates, nullptr, BB, 2048, 3072, 2048, 0);
    k_lstm_pw<<<(BB * 512 + 255) / 256, 256, 0, stream>>>(gates, atth, attc, atthbf);
    gemm(atthbf, wregh, reg_wh_b, qobj, nullptr, BB, 512, 512, 512, 0);
    gemm(atthbf, wfrmh, frm_wh_b, qfrm, nullptr, BB, 512, 512, 512, 0);
    k_attn<<<BB, 256, 0, stream>>>(object_feats, vobj, qobj, reg_wa, reg_wa_b, object_masks, NOO, 512,  cobj);
    k_attn<<<BB, 256, 0, stream>>>(frame_feats,  vfrm, qfrm, frm_wa, frm_wa_b, nullptr,      NFF, 1024, cfrm);
    k_pack_loc<<<BB, 256, 0, stream>>>(cobj, cfrm, locbf);
    gemm(locbf, wfuse, fuse_b, featv, nullptr, BB, 512, 1536, 512, 1);        // tanh
    k_pack3<<<BB, 256, 0, stream>>>(featv, atth, lngh /*=0*/, xh_lng);
    gemm(xh_lng, wlang, lang_b, gates, nullptr, BB, 2048, 1536, 2048, 0);
    k_lstm_pw<<<(BB * 512 + 255) / 256, 256, 0, stream>>>(gates, lngh, lngc, nullptr);

    // ---- 30 decode steps ----
    for (int t = 0; t < TT; ++t) {
        k_pack_att<<<BB, 256, 0, stream>>>(lngh, atth, gfeat, iword, word_embed, captions, t, xh_att);
        gemm(xh_att, watt, att_b, gates, nullptr, BB, 2048, 3072, 2048, 0);
        k_lstm_pw<<<(BB * 512 + 255) / 256, 256, 0, stream>>>(gates, atth, attc, atthbf);
        gemm(atthbf, wregh, reg_wh_b, qobj, nullptr, BB, 512, 512, 512, 0);
        gemm(atthbf, wfrmh, frm_wh_b, qfrm, nullptr, BB, 512, 512, 512, 0);
        k_attn<<<BB, 256, 0, stream>>>(object_feats, vobj, qobj, reg_wa, reg_wa_b, object_masks, NOO, 512,  cobj);
        k_attn<<<BB, 256, 0, stream>>>(frame_feats,  vfrm, qfrm, frm_wa, frm_wa_b, nullptr,      NFF, 1024, cfrm);
        k_pack_loc<<<BB, 256, 0, stream>>>(cobj, cfrm, locbf);
        gemm(locbf, wfuse, fuse_b, featv, nullptr, BB, 512, 1536, 512, 1);
        k_pack3<<<BB, 256, 0, stream>>>(featv, atth, lngh /*prev*/, xh_lng);
        gemm(xh_lng, wlang, lang_b, gates, nullptr, BB, 2048, 1536, 2048, 0);
        k_lstm_pw<<<(BB * 512 + 255) / 256, 256, 0, stream>>>(gates, lngh, lngc, nullptr);
        k_pack3<<<BB, 256, 0, stream>>>(lngh /*new*/, featv, atth, deccat);
        gemm(deccat, wout, out_b, decot, dobf, BB, 512, 1536, 512, 1);        // tanh + bf16 copy
        // vocab projection straight into d_out[:, t, :]  (ldc = T*V), 5 N-tiles/wave
        gemm5(dobf, wres, restore_b, out + (size_t)t * VV, BB, VV, 512, TT * VV);
    }
}
